// NeoTCNAttention_72198400246014
// MI455X (gfx1250) — matrix operand order repaired and checked by reading, not compiled, not run on hardware
//
#include <hip/hip_runtime.h>
#include <hip/hip_bf16.h>
#include <stdint.h>

#define HEADS 8
#define HD    64
#define SEQN  2048
#define NBATCH 4
#define EMBED 512
#define P_PERS 16
#define KTOT  2064      // 2048 + 16 persistent
#define KPAD  2080      // padded to 65 tiles of 32
#define NKT   65
#define SOFTSCALE 0.044194173824159216f  // 1/sqrt(512)

typedef __attribute__((ext_vector_type(16))) __bf16 v16bf;
typedef __attribute__((ext_vector_type(8)))  float  v8f;

union FragB { v16bf v; __bf16 b[16]; unsigned short u[16]; uint4 q[2]; };

// hardware bf16 convert (fptrunc, RNE) -> u16 bits
__device__ __forceinline__ unsigned short f2bfu(float f) {
  union { __bf16 b; unsigned short u; } cv;
  cv.b = (__bf16)f;
  return cv.u;
}
__device__ __forceinline__ v8f vzero8() {
  v8f z;
#pragma unroll
  for (int i = 0; i < 8; ++i) z[i] = 0.f;
  return z;
}

// ---------------------------------------------------------------- f32 -> bf16
__global__ void k_cvt_bf16(const float* __restrict__ src,
                           unsigned short* __restrict__ dst, int n) {
  int i = blockIdx.x * 256 + threadIdx.x;
  if (i < n) dst[i] = f2bfu(src[i]);
}

// ------------------------------------------------- per-head projection (X W^T)
// X: [N*SEQ, 512] f32, head h uses cols [64h,64h+64). Wb: [64,64] bf16, y = x W^T.
// Output bf16: row-major [head, seq, 64] (transposed==0) or [head, 64, KPAD].
__global__ __launch_bounds__(256) void k_proj(const float* __restrict__ X,
                                              const unsigned short* __restrict__ Wb,
                                              unsigned short* __restrict__ out,
                                              int headStride, int transposed) {
  const int wave = threadIdx.x >> 5, lane = threadIdx.x & 31;
  const int ln = lane & 15, hi = lane >> 4;
  const int h = wave;
  const int rowbase = blockIdx.x * 16;

  FragB a[2];
#pragma unroll
  for (int c = 0; c < 2; ++c) {
    const float* px = X + (size_t)(rowbase + ln) * EMBED + h * HD + c * 32 + hi * 8;
#pragma unroll
    for (int j = 0; j < 8; ++j) a[c].b[j] = (__bf16)px[j];
#pragma unroll
    for (int j = 0; j < 8; ++j) a[c].b[8 + j] = (__bf16)px[16 + j];
  }
#pragma unroll
  for (int ot = 0; ot < 4; ++ot) {
    v8f acc = vzero8();
#pragma unroll
    for (int c = 0; c < 2; ++c) {
      FragB b;
      const unsigned short* pw = Wb + (size_t)(ot * 16 + ln) * HD + c * 32 + hi * 8;
      b.q[0] = *(const uint4*)(pw);
      b.q[1] = *(const uint4*)(pw + 16);
      acc = __builtin_amdgcn_wmma_f32_16x16x32_bf16(false, a[c].v, false, b.v,
                                                    (short)0, acc, false, false);
    }
#pragma unroll
    for (int r = 0; r < 8; ++r) {
      int rg = rowbase + r + 8 * hi;
      int n = rg >> 11, lr = rg & 2047;
      int col = ot * 16 + ln;
      size_t base = (size_t)(n * HEADS + h) * (size_t)headStride;
      if (!transposed) out[base + (size_t)lr * HD + col] = f2bfu(acc[r]);
      else             out[base + (size_t)col * KPAD + lr] = f2bfu(acc[r]);
    }
  }
}

// ------------------------------------ append persistent k/v (+ zero pad rows)
__global__ void k_persist(const float* __restrict__ pk, const float* __restrict__ pv,
                          unsigned short* __restrict__ Kp,
                          unsigned short* __restrict__ Vt) {
  int i = blockIdx.x * 256 + threadIdx.x;  // N*H*32*64 = 65536
  int d = i & 63, p = (i >> 6) & 31, nh = i >> 11;
  float kv = (p < P_PERS) ? pk[p * HD + d] : 0.f;
  float vv = (p < P_PERS) ? pv[p * HD + d] : 0.f;
  Kp[(size_t)nh * KPAD * HD + (size_t)(SEQN + p) * HD + d] = f2bfu(kv);
  Vt[(size_t)nh * HD * KPAD + (size_t)d * KPAD + (SEQN + p)] = f2bfu(vv);
}

// --------------------------------------------------------- attention kernel
// block = 8 waves; wave h computes raw head-h energies; wave g owns mixed head g.
__global__ __launch_bounds__(256) void k_attn(
    const unsigned short* __restrict__ Qp, const unsigned short* __restrict__ Kp,
    const unsigned short* __restrict__ Vt, const unsigned char* __restrict__ mask,
    const float* __restrict__ Wpre, const float* __restrict__ Wpost,
    unsigned short* __restrict__ Ao) {
  __shared__ float bufE[HEADS][16][32];
  __shared__ float bufP[HEADS][16][32];
  __shared__ unsigned char mtile[16][32];

  const int wave = threadIdx.x >> 5, lane = threadIdx.x & 31;
  const int ln = lane & 15, hi = lane >> 4;
  const int n = blockIdx.x >> 7;
  const int qbase = (blockIdx.x & 127) * 16;

  const unsigned short* Qh = Qp + (size_t)(n * HEADS + wave) * SEQN * HD;
  const unsigned short* Kh = Kp + (size_t)(n * HEADS + wave) * KPAD * HD;
  const unsigned short* Vh = Vt + (size_t)(n * HEADS + wave) * HD * KPAD;

  FragB qa[2];
  {
    const unsigned short* pq = Qh + (size_t)(qbase + ln) * HD;
#pragma unroll
    for (int c = 0; c < 2; ++c) {
      qa[c].q[0] = *(const uint4*)(pq + c * 32 + hi * 8);
      qa[c].q[1] = *(const uint4*)(pq + c * 32 + hi * 8 + 16);
    }
  }
  float wpre[8], wpost[8];
#pragma unroll
  for (int j = 0; j < 8; ++j) { wpre[j] = Wpre[wave * 8 + j]; wpost[j] = Wpost[wave * 8 + j]; }
  const float slope = exp2f(-1.25f * (float)(wave + 2));  // 2^(-alpha*(h+1+start_i))

  float m[8], s[8], invs[8];
#pragma unroll
  for (int r = 0; r < 8; ++r) { m[r] = -3.0e38f; s[r] = 0.f; invs[r] = 0.f; }

  v8f oacc[4];
#pragma unroll
  for (int dt = 0; dt < 4; ++dt) oacc[dt] = vzero8();

  for (int pass = 0; pass < 2; ++pass) {
    if (pass == 1) {
#pragma unroll
      for (int r = 0; r < 8; ++r) invs[r] = (s[r] > 0.f) ? (1.f / s[r]) : 0.f;
    }
    for (int t = 0; t < NKT; ++t) {
      const int kbase = t * 32;
      // stage 0: wave 0 stages the mask tile (persistent/pad region == valid)
      if (wave == 0) {
        int rowm = lane >> 1, ch = lane & 1;
        uint4 mv;
        if (kbase < SEQN)
          mv = *(const uint4*)(mask + ((size_t)n * SEQN + qbase + rowm) * SEQN + kbase + ch * 16);
        else
          mv = make_uint4(0x01010101u, 0x01010101u, 0x01010101u, 0x01010101u);
        *(uint4*)(&mtile[rowm][ch * 16]) = mv;
      }
      // stage 1: raw energies for head `wave` + ALiBi, into LDS
#pragma unroll
      for (int st = 0; st < 2; ++st) {
        v8f acc = vzero8();
#pragma unroll
        for (int c = 0; c < 2; ++c) {
          FragB b;
          const unsigned short* pk2 =
              Kh + (size_t)(kbase + st * 16 + ln) * HD + c * 32 + hi * 8;
          b.q[0] = *(const uint4*)(pk2);
          b.q[1] = *(const uint4*)(pk2 + 16);
          acc = __builtin_amdgcn_wmma_f32_16x16x32_bf16(false, qa[c].v, false, b.v,
                                                        (short)0, acc, false, false);
        }
        const int kcol = kbase + st * 16 + ln;
#pragma unroll
        for (int r = 0; r < 8; ++r) {
          float e = acc[r];
          if (kcol < SEQN)
            e -= fabsf((float)(qbase + r + 8 * hi - kcol)) * slope;
          bufE[wave][r + 8 * hi][st * 16 + ln] = e;
        }
      }
      __syncthreads();
      // stage 2: pre-softmax talking-heads mix, mask, scale -> logits
      float l[16];
#pragma unroll
      for (int st = 0; st < 2; ++st) {
        const int kcol = kbase + st * 16 + ln;
#pragma unroll
        for (int r = 0; r < 8; ++r) {
          float sum = 0.f;
#pragma unroll
          for (int h2 = 0; h2 < 8; ++h2)
            sum += wpre[h2] * bufE[h2][r + 8 * hi][st * 16 + ln];
          float lv;
          if (kcol >= KTOT) lv = -1.0e30f;   // pad -> excluded
          else {
            if (kcol < SEQN && mtile[r + 8 * hi][st * 16 + ln] == 0) sum = -1.0e4f;
            lv = sum * SOFTSCALE;
          }
          l[st * 8 + r] = lv;
        }
      }
      if (pass == 0) {  // online row max / sum (reduce across 16 lanes per half)
#pragma unroll
        for (int r = 0; r < 8; ++r) {
          float v = fmaxf(l[r], l[8 + r]);
          v = fmaxf(v, __shfl_xor(v, 1, 16));
          v = fmaxf(v, __shfl_xor(v, 2, 16));
          v = fmaxf(v, __shfl_xor(v, 4, 16));
          v = fmaxf(v, __shfl_xor(v, 8, 16));
          float mn = fmaxf(m[r], v);
          float es = __expf(l[r] - mn) + __expf(l[8 + r] - mn);
          es += __shfl_xor(es, 1, 16);
          es += __shfl_xor(es, 2, 16);
          es += __shfl_xor(es, 4, 16);
          es += __shfl_xor(es, 8, 16);
          s[r] = s[r] * __expf(m[r] - mn) + es;
          m[r] = mn;
        }
        __syncthreads();
      } else {
        // normalized probabilities for mixed head `wave`
#pragma unroll
        for (int st = 0; st < 2; ++st)
#pragma unroll
          for (int r = 0; r < 8; ++r)
            bufP[wave][r + 8 * hi][st * 16 + ln] =
                __expf(l[st * 8 + r] - m[r]) * invs[r];
        __syncthreads();
        // stage 3: post-softmax mix -> bf16 A fragment, then PV WMMAs
        FragB pa;
#pragma unroll
        for (int j = 0; j < 16; ++j) {
          int kpos = ((j < 8) ? j : j + 8) + 8 * hi;  // A-layout K index
          float sum = 0.f;
#pragma unroll
          for (int h2 = 0; h2 < 8; ++h2) sum += wpost[h2] * bufP[h2][ln][kpos];
          pa.b[j] = (__bf16)sum;
        }
#pragma unroll
        for (int dt = 0; dt < 4; ++dt) {
          FragB b;
          const unsigned short* pv2 =
              Vh + (size_t)(dt * 16 + ln) * KPAD + kbase + hi * 8;
          b.q[0] = *(const uint4*)(pv2);
          b.q[1] = *(const uint4*)(pv2 + 16);
          oacc[dt] = __builtin_amdgcn_wmma_f32_16x16x32_bf16(
              false, pa.v, false, b.v, (short)0, oacc[dt], false, false);
        }
        __syncthreads();
      }
    }
  }
  // store attention output (bf16, [N*SEQ, 512], head-major columns)
#pragma unroll
  for (int dt = 0; dt < 4; ++dt)
#pragma unroll
    for (int r = 0; r < 8; ++r) {
      size_t rowg = (size_t)n * SEQN + qbase + r + 8 * hi;
      Ao[rowg * EMBED + wave * HD + dt * 16 + ln] = f2bfu(oacc[dt][r]);
    }
}

// ------------------------------------------------------------- final FC + bias
__global__ __launch_bounds__(256) void k_fc(const unsigned short* __restrict__ Ao,
                                            const unsigned short* __restrict__ Wf,
                                            const float* __restrict__ bias,
                                            float* __restrict__ out) {
  const int wave = threadIdx.x >> 5, lane = threadIdx.x & 31;
  const int ln = lane & 15, hi = lane >> 4;
  const int wg = blockIdx.x * 8 + wave;     // 16384 waves
  const int rowbase = (wg >> 5) * 16;
  const int ot = wg & 31;
  v8f acc = vzero8();
  const unsigned short* pa0 = Ao + (size_t)(rowbase + ln) * EMBED;
  const unsigned short* pb0 = Wf + (size_t)(ot * 16 + ln) * EMBED;
  for (int ec = 0; ec < 16; ++ec) {
    FragB a, b;
    a.q[0] = *(const uint4*)(pa0 + ec * 32 + hi * 8);
    a.q[1] = *(const uint4*)(pa0 + ec * 32 + hi * 8 + 16);
    b.q[0] = *(const uint4*)(pb0 + ec * 32 + hi * 8);
    b.q[1] = *(const uint4*)(pb0 + ec * 32 + hi * 8 + 16);
    acc = __builtin_amdgcn_wmma_f32_16x16x32_bf16(false, a.v, false, b.v,
                                                  (short)0, acc, false, false);
  }
  float bv = bias[ot * 16 + ln];
#pragma unroll
  for (int r = 0; r < 8; ++r)
    out[(size_t)(rowbase + r + 8 * hi) * EMBED + ot * 16 + ln] = acc[r] + bv;
}

extern "C" void kernel_launch(void* const* d_in, const int* in_sizes, int n_in,
                              void* d_out, int out_size, void* d_ws, size_t ws_size,
                              hipStream_t stream) {
  const float* values  = (const float*)d_in[0];
  const float* keys    = (const float*)d_in[1];
  const float* queries = (const float*)d_in[2];
  const unsigned char* mask = (const unsigned char*)d_in[3];
  const float* Wv    = (const float*)d_in[4];
  const float* Wk    = (const float*)d_in[5];
  const float* Wq    = (const float*)d_in[6];
  const float* Wpre  = (const float*)d_in[7];
  const float* Wpost = (const float*)d_in[8];
  const float* pkeys = (const float*)d_in[9];
  const float* pvals = (const float*)d_in[10];
  const float* fcw   = (const float*)d_in[11];
  const float* fcb   = (const float*)d_in[12];
  float* out = (float*)d_out;

  char* ws = (char*)d_ws;
  const size_t szQ  = (size_t)NBATCH * HEADS * SEQN * HD * 2;   // 8,388,608
  const size_t szK  = (size_t)NBATCH * HEADS * KPAD * HD * 2;   // 8,519,680
  const size_t szV  = szK;
  const size_t szAo = (size_t)NBATCH * SEQN * EMBED * 2;        // 8,388,608
  const size_t szWf = (size_t)EMBED * EMBED * 2;                // 524,288
  const size_t szWs = (size_t)HD * HD * 2;                      // 8,192 each
  unsigned short* Qp  = (unsigned short*)(ws);
  unsigned short* Kp  = (unsigned short*)(ws + szQ);
  unsigned short* Vt  = (unsigned short*)(ws + szQ + szK);
  unsigned short* Ao  = (unsigned short*)(ws + szQ + szK + szV);
  unsigned short* Wf  = (unsigned short*)(ws + szQ + szK + szV + szAo);
  unsigned short* Wqb = (unsigned short*)(ws + szQ + szK + szV + szAo + szWf);
  unsigned short* Wkb = (unsigned short*)(ws + szQ + szK + szV + szAo + szWf + szWs);
  unsigned short* Wvb = (unsigned short*)(ws + szQ + szK + szV + szAo + szWf + 2 * szWs);

  k_cvt_bf16<<<1024, 256, 0, stream>>>(fcw, Wf, EMBED * EMBED);
  k_cvt_bf16<<<16, 256, 0, stream>>>(Wq, Wqb, HD * HD);
  k_cvt_bf16<<<16, 256, 0, stream>>>(Wk, Wkb, HD * HD);
  k_cvt_bf16<<<16, 256, 0, stream>>>(Wv, Wvb, HD * HD);
  k_proj<<<512, 256, 0, stream>>>(queries, Wqb, Qp, SEQN * HD, 0);
  k_proj<<<512, 256, 0, stream>>>(keys,    Wkb, Kp, KPAD * HD, 0);
  k_proj<<<512, 256, 0, stream>>>(values,  Wvb, Vt, HD * KPAD, 1);
  k_persist<<<256, 256, 0, stream>>>(pkeys, pvals, Kp, Vt);
  k_attn<<<512, 256, 0, stream>>>(Qp, Kp, Vt, mask, Wpre, Wpost, Ao);
  k_fc<<<2048, 256, 0, stream>>>(Ao, Wf, fcb, out);
}
